// MultiHeadAttention_67920612819294
// MI455X (gfx1250) — matrix operand order repaired and checked by reading, not compiled, not run on hardware
//
#include <hip/hip_runtime.h>
#include <hip/hip_bf16.h>
#include <cstdint>

typedef __attribute__((ext_vector_type(16))) _Float16 v16h;
typedef __attribute__((ext_vector_type(8)))  float    v8f;
typedef __attribute__((ext_vector_type(4)))  uint32_t u32x4;
typedef __attribute__((ext_vector_type(8)))  uint32_t u32x8;

#define D_MODEL 1024
#define NUM_HEADS 16
#define D_K 64
#define SEQ 2048
#define BATCH 4
#define MTOT (BATCH * SEQ)          // 8192
#define BH (BATCH * NUM_HEADS)      // 64

// LDS leading dims (f16 units), padded for conflict-free gathers
#define LDA32 34   // 32-wide K tiles (stride 68B -> 17 banks, gcd(17,64)=1)
#define LDK64 72   // 64-wide d tiles (stride 144B -> 36 banks, 16 distinct)

// ---------------- WMMA fragment helpers (wave32, ISA 7.12.2 layouts) --------

__device__ __forceinline__ v8f wmma_f16(v16h a, v16h b, v8f c) {
  return __builtin_amdgcn_wmma_f32_16x16x32_f16(
      /*neg_a=*/false, a, /*neg_b=*/false, b,
      /*c_mod=*/(short)0, c, /*reuse_a=*/false, /*reuse_b=*/false);
}

// A: 16(M) x 32(K), LDS row-major [M][ld]
__device__ __forceinline__ v16h load_a(const _Float16* s, int ld) {
  int lane = threadIdx.x & 31;
  int m = lane & 15;
  int koff = (lane & 16) >> 1;            // 0 or 8
  v16h a;
#pragma unroll
  for (int e = 0; e < 16; ++e) {
    int k = (koff << 1) + e;                   // contiguous: 0-15 / 16-31
    a[e] = s[m * ld + k];
  }
  return a;
}

// B: 32(K) x 16(N), LDS holds element (k,n) at s[n*ld + k]  ("NK": W rows / K rows)
__device__ __forceinline__ v16h load_b_nk(const _Float16* s, int ld) {
  int lane = threadIdx.x & 31;
  int n = lane & 15;
  int kbase = lane & 16;                  // 0 or 16
  v16h b;
#pragma unroll
  for (int e = 0; e < 16; ++e) b[e] = s[n * ld + kbase + e];
  return b;
}

// B: 32(K) x 16(N), LDS holds element (k,n) at s[k*ld + n]  ("KN": V tiles)
__device__ __forceinline__ v16h load_b_kn(const _Float16* s, int ld) {
  int lane = threadIdx.x & 31;
  int n = lane & 15;
  int kbase = lane & 16;
  v16h b;
#pragma unroll
  for (int e = 0; e < 16; ++e) b[e] = s[(kbase + e) * ld + n];
  return b;
}

// ---------------- TDM: 1-D contiguous global -> LDS with row padding -------
// Copies `elems` f16 (contiguous in global) into LDS, inserting 4 DWORDs of
// pad after every 32 DWORDs stored (64-f16 rows -> 72-f16 LDS stride = LDK64).
// D# per ISA 8.3/8.4. Issue from ONE wave; wait TENSORcnt; then barrier.
__device__ __forceinline__ void tdm_load_1d_pad(uint32_t lds_off,
                                                const void* gptr,
                                                uint32_t elems) {
  uint64_t ga = (uint64_t)(uintptr_t)gptr;
  u32x4 g0;
  g0[0] = 1u;                                        // count=1, user mode
  g0[1] = lds_off;                                   // lds_addr (bytes)
  g0[2] = (uint32_t)ga;                              // global_addr[31:0]
  g0[3] = (uint32_t)((ga >> 32) & 0x1FFFFFFu)        // global_addr[56:32]
          | (2u << 30);                              // type=2 ("image")
  u32x8 g1;
  g1[0] = (1u << 16)                                 // data_size = 2 bytes
        | (1u << 20)                                 // pad_enable
        | (4u << 22)                                 // pad_interval: 32 DWORDs
        | (3u << 25);                                // pad_amount:   4 DWORDs
  g1[1] = (elems & 0xFFFFu) << 16;                   // tensor_dim0[15:0]
  g1[2] = (elems >> 16);                             // tensor_dim0[31:16]
  g1[3] = (elems & 0xFFFFu) << 16;                   // tile_dim0 (1-D tile)
  g1[4] = 0u;                                        // tile_dim1=0 (unused)
  g1[5] = elems;                                     // tensor_dim0_stride lo
  g1[6] = 0u;
  g1[7] = 0u;
  asm volatile("tensor_load_to_lds %0, %1" : : "s"(g0), "s"(g1) : "memory");
}

__device__ __forceinline__ uint32_t lds_offset(const void* p) {
  return (uint32_t)(uintptr_t)p;   // generic shared ptr: low 32 bits = LDS off
}

// ---------------- Kernel 1: fused QKV projection ---------------------------
// Y[m,n] = sum_k X[m,k] * W[n,k] + bias[n]; store f16 in (B,H,S,d_k) layout.
__global__ __launch_bounds__(256) void k_qkv(
    const float* __restrict__ q, const float* __restrict__ k,
    const float* __restrict__ v,
    const float* __restrict__ wq, const float* __restrict__ wk,
    const float* __restrict__ wv,
    const float* __restrict__ bq, const float* __restrict__ bk,
    const float* __restrict__ bv,
    _Float16* __restrict__ Qh, _Float16* __restrict__ Kh,
    _Float16* __restrict__ Vh) {
  const float* X; const float* W; const float* B; _Float16* O;
  if (blockIdx.z == 0)      { X = q; W = wq; B = bq; O = Qh; }
  else if (blockIdx.z == 1) { X = k; W = wk; B = bk; O = Kh; }
  else                      { X = v; W = wv; B = bv; O = Vh; }

  __shared__ _Float16 Xs[64 * LDA32];
  __shared__ _Float16 Ws[64 * LDA32];

  const int tid = threadIdx.x;
  const int wave = tid >> 5;
  const int M0 = blockIdx.y * 64;
  const int N0 = blockIdx.x * 64;
  const int mi = wave & 3;
  const int ni0 = (wave >> 2) * 2;

  v8f acc0 = {}; v8f acc1 = {};

  for (int k0 = 0; k0 < D_MODEL; k0 += 32) {
#pragma unroll
    for (int i = 0; i < 8; ++i) {
      int e = tid + i * 256;
      int r = e >> 5, c = e & 31;
      Xs[r * LDA32 + c] = (_Float16)X[(size_t)(M0 + r) * D_MODEL + k0 + c];
      Ws[r * LDA32 + c] = (_Float16)W[(size_t)(N0 + r) * D_MODEL + k0 + c];
    }
    __syncthreads();
    v16h a  = load_a(&Xs[mi * 16 * LDA32], LDA32);
    v16h b0 = load_b_nk(&Ws[(ni0 * 16) * LDA32], LDA32);
    v16h b1 = load_b_nk(&Ws[((ni0 + 1) * 16) * LDA32], LDA32);
    acc0 = wmma_f16(a, b0, acc0);
    acc1 = wmma_f16(a, b1, acc1);
    __syncthreads();
  }

  const int lane = tid & 31;
  const int nn = lane & 15;
  const int mo = (lane & 16) >> 1;
#pragma unroll
  for (int j = 0; j < 2; ++j) {
    const v8f& acc = j ? acc1 : acc0;
    int n = N0 + (ni0 + j) * 16 + nn;
    float bias = B[n];
    int h = n >> 6, d = n & 63;
#pragma unroll
    for (int e = 0; e < 8; ++e) {
      int m = M0 + mi * 16 + mo + e;
      int bb = m >> 11, s = m & 2047;
      O[(((size_t)bb * NUM_HEADS + h) * SEQ + s) * D_K + d] =
          (_Float16)(acc[e] + bias);
    }
  }
}

// ---------------- Kernel 2: scores = (Qh Kh^T) / 8  -> raw fp32 ------------
// Q/K tiles are contiguous f16 blocks in global -> staged by TDM with padding.
__global__ __launch_bounds__(256) void k_scores(
    const _Float16* __restrict__ Qh, const _Float16* __restrict__ Kh,
    float* __restrict__ attn) {
  const int bh = blockIdx.y;
  const int q0 = blockIdx.x * 64;
  const _Float16* Qb = Qh + (size_t)bh * SEQ * D_K;
  const _Float16* Kb = Kh + (size_t)bh * SEQ * D_K;
  float* Ab = attn + (size_t)bh * SEQ * SEQ;

  __shared__ _Float16 Qs[64 * LDK64];
  __shared__ _Float16 Ks[128 * LDK64];

  const int tid = threadIdx.x;
  const int wave = tid >> 5;
  const int mi = wave & 3;
  const int wcol = wave >> 2;
  const int lane = tid & 31;
  const int nn = lane & 15;
  const int mo = (lane & 16) >> 1;

  if (tid < 32) {                       // wave 0 drives the TDM
    tdm_load_1d_pad(lds_offset(Qs), Qb + (size_t)q0 * D_K, 64 * D_K);
    __builtin_amdgcn_s_wait_tensorcnt(0);
  }
  __syncthreads();
  v16h a0 = load_a(&Qs[mi * 16 * LDK64], LDK64);       // d = 0..31
  v16h a1 = load_a(&Qs[mi * 16 * LDK64 + 32], LDK64);  // d = 32..63

  for (int s0 = 0; s0 < SEQ; s0 += 128) {
    __syncthreads();                    // previous Ks tile fully consumed
    if (tid < 32) {
      tdm_load_1d_pad(lds_offset(Ks), Kb + (size_t)s0 * D_K, 128 * D_K);
      __builtin_amdgcn_s_wait_tensorcnt(0);
    }
    __syncthreads();
#pragma unroll
    for (int ci = 0; ci < 4; ++ci) {
      int ncol = wcol * 64 + ci * 16;
      v16h b0 = load_b_nk(&Ks[ncol * LDK64], LDK64);
      v16h b1 = load_b_nk(&Ks[ncol * LDK64 + 32], LDK64);
      v8f acc = {};
      acc = wmma_f16(a0, b0, acc);
      acc = wmma_f16(a1, b1, acc);
      float* dst = Ab + (size_t)(q0 + mi * 16) * SEQ + s0 + ncol;
#pragma unroll
      for (int e = 0; e < 8; ++e)
        dst[(size_t)(mo + e) * SEQ + nn] = acc[e] * 0.125f;  // 1/sqrt(64)
    }
  }
}

// ---------------- Kernel 3: row softmax in place ---------------------------
__global__ __launch_bounds__(256) void k_softmax(float* __restrict__ attn) {
  __shared__ float red[256];
  const size_t row = blockIdx.x;            // B*H*S rows
  float* p = attn + row * (size_t)SEQ;
  const int tid = threadIdx.x;

  float v[8];
  float m = -3.402823466e38f;
#pragma unroll
  for (int i = 0; i < 8; ++i) { v[i] = p[tid + i * 256]; m = fmaxf(m, v[i]); }
  red[tid] = m; __syncthreads();
  for (int s = 128; s > 0; s >>= 1) {
    if (tid < s) red[tid] = fmaxf(red[tid], red[tid + s]);
    __syncthreads();
  }
  m = red[0]; __syncthreads();

  float sum = 0.f;
#pragma unroll
  for (int i = 0; i < 8; ++i) { v[i] = __expf(v[i] - m); sum += v[i]; }
  red[tid] = sum; __syncthreads();
  for (int s = 128; s > 0; s >>= 1) {
    if (tid < s) red[tid] += red[tid + s];
    __syncthreads();
  }
  float inv = 1.0f / red[0];
#pragma unroll
  for (int i = 0; i < 8; ++i) p[tid + i * 256] = v[i] * inv;
}

// ---------------- Kernel 4: x = attn @ V  (f16 out, (B,H,S,d_k)) -----------
// V tile staged by TDM (overlaps with all-thread fp32->f16 attn staging).
__global__ __launch_bounds__(256) void k_av(
    const float* __restrict__ attn, const _Float16* __restrict__ Vh,
    _Float16* __restrict__ Xh) {
  const int bh = blockIdx.y;
  const int q0 = blockIdx.x * 64;
  const float* Ab = attn + (size_t)bh * SEQ * SEQ;
  const _Float16* Vb = Vh + (size_t)bh * SEQ * D_K;
  _Float16* Xb = Xh + (size_t)bh * SEQ * D_K;

  __shared__ _Float16 As[64 * LDA32];
  __shared__ _Float16 Vs[32 * LDK64];

  const int tid = threadIdx.x;
  const int wave = tid >> 5;
  const int mi = wave & 3;
  const int ni0 = (wave >> 2) * 2;

  v8f acc0 = {}; v8f acc1 = {};

  for (int s0 = 0; s0 < SEQ; s0 += 32) {
    __syncthreads();                    // previous As/Vs fully consumed
    if (tid < 32)                       // async DMA of V tile
      tdm_load_1d_pad(lds_offset(Vs), Vb + (size_t)s0 * D_K, 32 * D_K);
#pragma unroll
    for (int i = 0; i < 8; ++i) {       // 64x32 attn tile, fp32 -> f16
      int e = tid + i * 256;
      int r = e >> 5, c = e & 31;
      As[r * LDA32 + c] = (_Float16)Ab[(size_t)(q0 + r) * SEQ + s0 + c];
    }
    if (tid < 32) __builtin_amdgcn_s_wait_tensorcnt(0);
    __syncthreads();
    v16h a  = load_a(&As[mi * 16 * LDA32], LDA32);
    v16h b0 = load_b_kn(&Vs[ni0 * 16], LDK64);
    v16h b1 = load_b_kn(&Vs[(ni0 + 1) * 16], LDK64);
    acc0 = wmma_f16(a, b0, acc0);
    acc1 = wmma_f16(a, b1, acc1);
  }

  const int lane = tid & 31;
  const int nn = lane & 15;
  const int mo = (lane & 16) >> 1;
#pragma unroll
  for (int j = 0; j < 2; ++j) {
    const v8f& acc = j ? acc1 : acc0;
    int d = (ni0 + j) * 16 + nn;
#pragma unroll
    for (int e = 0; e < 8; ++e)
      Xb[(size_t)(q0 + mi * 16 + mo + e) * D_K + d] = (_Float16)acc[e];
  }
}

// ---------------- Kernel 5: out = x @ w_o^T + b_o --------------------------
__global__ __launch_bounds__(256) void k_out(
    const _Float16* __restrict__ Xh, const float* __restrict__ wo,
    const float* __restrict__ bo, float* __restrict__ out) {
  __shared__ _Float16 Xs[64 * LDA32];
  __shared__ _Float16 Ws[64 * LDA32];

  const int tid = threadIdx.x;
  const int wave = tid >> 5;
  const int M0 = blockIdx.y * 64;
  const int N0 = blockIdx.x * 64;
  const int mi = wave & 3;
  const int ni0 = (wave >> 2) * 2;

  v8f acc0 = {}; v8f acc1 = {};

  for (int k0 = 0; k0 < D_MODEL; k0 += 32) {
    const int h = k0 >> 6, d0 = k0 & 63;   // 32-chunk lies inside one head
#pragma unroll
    for (int i = 0; i < 8; ++i) {
      int e = tid + i * 256;
      int r = e >> 5, c = e & 31;
      int m = M0 + r;
      int bb = m >> 11, s = m & 2047;
      Xs[r * LDA32 + c] =
          Xh[(((size_t)bb * NUM_HEADS + h) * SEQ + s) * D_K + d0 + c];
      Ws[r * LDA32 + c] = (_Float16)wo[(size_t)(N0 + r) * D_MODEL + k0 + c];
    }
    __syncthreads();
    v16h a  = load_a(&Xs[mi * 16 * LDA32], LDA32);
    v16h b0 = load_b_nk(&Ws[(ni0 * 16) * LDA32], LDA32);
    v16h b1 = load_b_nk(&Ws[((ni0 + 1) * 16) * LDA32], LDA32);
    acc0 = wmma_f16(a, b0, acc0);
    acc1 = wmma_f16(a, b1, acc1);
    __syncthreads();
  }

  const int lane = tid & 31;
  const int nn = lane & 15;
  const int mo = (lane & 16) >> 1;
#pragma unroll
  for (int j = 0; j < 2; ++j) {
    const v8f& acc = j ? acc1 : acc0;
    int n = N0 + (ni0 + j) * 16 + nn;
    float bias = bo[n];
#pragma unroll
    for (int e = 0; e < 8; ++e)
      out[(size_t)(M0 + mi * 16 + mo + e) * D_MODEL + n] = acc[e] + bias;
  }
}

// ---------------- Launch ----------------------------------------------------
extern "C" void kernel_launch(void* const* d_in, const int* in_sizes, int n_in,
                              void* d_out, int out_size, void* d_ws,
                              size_t ws_size, hipStream_t stream) {
  (void)in_sizes; (void)n_in; (void)out_size; (void)ws_size;

  const float* q  = (const float*)d_in[0];
  const float* k  = (const float*)d_in[1];
  const float* v  = (const float*)d_in[2];
  const float* wq = (const float*)d_in[3];
  const float* bq = (const float*)d_in[4];
  const float* wk = (const float*)d_in[5];
  const float* bk = (const float*)d_in[6];
  const float* wv = (const float*)d_in[7];
  const float* bv = (const float*)d_in[8];
  const float* wo = (const float*)d_in[9];
  const float* bo = (const float*)d_in[10];

  float* out  = (float*)d_out;                                  // (B,S,D)
  float* attn = out + (size_t)BATCH * SEQ * D_MODEL;            // (B,H,S,S)

  const size_t headElems = (size_t)BH * SEQ * D_K;              // 8,388,608
  _Float16* Qh = (_Float16*)d_ws;
  _Float16* Kh = Qh + headElems;
  _Float16* Vh = Kh + headElems;
  _Float16* Xh = Vh + headElems;

  dim3 blk(256);
  k_qkv<<<dim3(D_MODEL / 64, MTOT / 64, 3), blk, 0, stream>>>(
      q, k, v, wq, wk, wv, bq, bk, bv, Qh, Kh, Vh);
  k_scores<<<dim3(SEQ / 64, BH), blk, 0, stream>>>(Qh, Kh, attn);
  k_softmax<<<dim3(BH * SEQ), blk, 0, stream>>>(attn);
  k_av<<<dim3(SEQ / 64, BH), blk, 0, stream>>>(attn, Vh, Xh);
  k_out<<<dim3(D_MODEL / 64, MTOT / 64), blk, 0, stream>>>(Xh, wo, bo, out);
}
